// LightGCN_40046275068578
// MI455X (gfx1250) — compile-verified, matrix operand order checked by reading, not run on hardware
//
#include <hip/hip_runtime.h>
#include <hip/hip_bf16.h>

#define NUM_USERS 100000
#define NUM_ITEMS 50000
#define NN        (NUM_USERS + NUM_ITEMS)
#define EMB       64
#define NF        ((long long)NN * EMB)     /* 9,600,000 floats = 38.4 MB */
#define N_EDGES   4800000
#define BATCH     4096

typedef __attribute__((ext_vector_type(2)))  float    v2f;
typedef __attribute__((ext_vector_type(8)))  float    v8f;
typedef __attribute__((ext_vector_type(16))) _Float16 v16h;

#if __has_builtin(__builtin_amdgcn_wmma_f32_16x16x4_f32)
#define HAVE_WMMA_F32X4 1
#else
#define HAVE_WMMA_F32X4 0
#endif

// ---------------------------------------------------------------------------
// Kernel 1: acc = X = concat(user_emb, item_emb), vectorized float4
// ---------------------------------------------------------------------------
__global__ void lgcn_init(const float* __restrict__ ue, const float* __restrict__ ie,
                          float* __restrict__ acc, float* __restrict__ x) {
  long long i = ((long long)blockIdx.x * blockDim.x + threadIdx.x) * 4;
  if (i >= NF) return;
  const long long UF = (long long)NUM_USERS * EMB;  // multiple of 4, no straddle
  float4 v;
  if (i < UF) v = *(const float4*)(ue + i);
  else        v = *(const float4*)(ie + (i - UF));
  *(float4*)(acc + i) = v;
  *(float4*)(x   + i) = v;
}

// ---------------------------------------------------------------------------
// Kernel 2: SpMM scatter.  16 lanes per edge, one float4 per lane:
//   gather x[col] is a fully coalesced 256B line per edge; scatter via
//   hardware fp32 no-return atomics into L2 (table is L2-resident).
// ---------------------------------------------------------------------------
__global__ void lgcn_spmm(const int* __restrict__ rows, const int* __restrict__ cols,
                          const float* __restrict__ vals, const float* __restrict__ x,
                          float* __restrict__ y) {
  long long gid = (long long)blockIdx.x * blockDim.x + threadIdx.x;
  long long e   = gid >> 4;
  if (e >= N_EDGES) return;
  int sub = (int)(gid & 15);

  if (sub == 0 && e + 8192 < N_EDGES) {          // stream-ahead prefetch (global_prefetch_b8)
    __builtin_prefetch(rows + e + 8192, 0, 0);
    __builtin_prefetch(cols + e + 8192, 0, 0);
    __builtin_prefetch(vals + e + 8192, 0, 0);
  }

  int   r = rows[e];
  int   c = cols[e];
  float v = vals[e];

  float4 xv = *(const float4*)(x + (long long)c * EMB + sub * 4);
  float* yp = y + (long long)r * EMB + sub * 4;
  __hip_atomic_fetch_add(yp + 0, xv.x * v, __ATOMIC_RELAXED, __HIP_MEMORY_SCOPE_AGENT);
  __hip_atomic_fetch_add(yp + 1, xv.y * v, __ATOMIC_RELAXED, __HIP_MEMORY_SCOPE_AGENT);
  __hip_atomic_fetch_add(yp + 2, xv.z * v, __ATOMIC_RELAXED, __HIP_MEMORY_SCOPE_AGENT);
  __hip_atomic_fetch_add(yp + 3, xv.w * v, __ATOMIC_RELAXED, __HIP_MEMORY_SCOPE_AGENT);
}

// ---------------------------------------------------------------------------
// Kernel 3: acc += y (layer accumulation), vectorized float4
// ---------------------------------------------------------------------------
__global__ void lgcn_add(float* __restrict__ acc, const float* __restrict__ y) {
  long long i = ((long long)blockIdx.x * blockDim.x + threadIdx.x) * 4;
  if (i >= NF) return;
  float4 a = *(const float4*)(acc + i);
  float4 b = *(const float4*)(y + i);
  a.x += b.x; a.y += b.y; a.z += b.z; a.w += b.w;
  *(float4*)(acc + i) = a;
}

__device__ __forceinline__ float v8f_get(v8f v, int i) {
  switch (i & 7) {
    case 0: return v[0]; case 1: return v[1]; case 2: return v[2]; case 3: return v[3];
    case 4: return v[4]; case 5: return v[5]; case 6: return v[6]; default: return v[7];
  }
}

// ---------------------------------------------------------------------------
// Kernel 4: batched scoring via WMMA.  One wave per tile of 16 batch rows:
//   D = U (16x64) x P^T (64x16) accumulated as 16 x V_WMMA_F32_16X16X4_F32;
//   the 16 diagonal lanes emit the scores.  1/4 layer-average folded into a
//   final 1/16 scale on the dot product.
// ---------------------------------------------------------------------------
__global__ void lgcn_score(const float* __restrict__ acc,
                           const int* __restrict__ users,
                           const int* __restrict__ pos_items,
                           const int* __restrict__ neg_items,
                           float* __restrict__ out) {
  int wave = (int)((blockIdx.x * blockDim.x + threadIdx.x) >> 5);
  int lane = threadIdx.x & 31;
  int tile = wave;
  if (tile * 16 >= BATCH) return;              // uniform per wave -> EXEC stays all-1s

  int m  = lane & 15;                           // row/col index handled by this lane
  int bi = tile * 16 + m;
  const float* urow = acc + (long long)users[bi] * EMB;
  const float* prow = acc + ((long long)NUM_USERS + pos_items[bi]) * EMB;
  const float* nrow = acc + ((long long)NUM_USERS + neg_items[bi]) * EMB;

  v8f cp = {}; v8f cn = {};

#if HAVE_WMMA_F32X4
  int khalf = (lane >> 4) * 2;                  // lanes 0-15: K{0,1}; lanes 16-31: K{2,3}
  #pragma unroll
  for (int kb = 0; kb < 16; ++kb) {
    int k0 = kb * 4 + khalf;
    v2f a;  a.x  = urow[k0]; a.y  = urow[k0 + 1];
    v2f bp; bp.x = prow[k0]; bp.y = prow[k0 + 1];
    v2f bn; bn.x = nrow[k0]; bn.y = nrow[k0 + 1];
    cp = __builtin_amdgcn_wmma_f32_16x16x4_f32(false, a, false, bp, (short)0, cp, false, false);
    cn = __builtin_amdgcn_wmma_f32_16x16x4_f32(false, a, false, bn, (short)0, cn, false, false);
  }
#else
  // Fallback: fp16 inputs, fp32 accumulate, K=64 as two 16x16x32 WMMAs.
  int hi = (lane >> 4) * 8;
  #pragma unroll
  for (int kb = 0; kb < 2; ++kb) {
    v16h a, bp, bn;
    #pragma unroll
    for (int h = 0; h < 16; ++h) {
      int k = kb * 32 + hi + h + ((h >= 8) ? 8 : 0);
      a[h]  = (_Float16)urow[k];
      bp[h] = (_Float16)prow[k];
      bn[h] = (_Float16)nrow[k];
    }
    cp = __builtin_amdgcn_wmma_f32_16x16x32_f16(false, a, false, bp, (short)0, cp, false, false);
    cn = __builtin_amdgcn_wmma_f32_16x16x32_f16(false, a, false, bn, (short)0, cn, false, false);
  }
#endif

  // Diagonal extraction: C/D layout -> lane half (lane>>4) holds M = r + 8*(lane>>4),
  // N = lane&15.  Diagonal element m==n lives in lane where (lane>>4)==(n>>3), VGPR n&7.
  int n = lane & 15;
  if ((lane >> 4) == (n >> 3)) {
    float sp = v8f_get(cp, n & 7) * 0.0625f;    // (1/4)^2 from final = acc/4 on both sides
    float sn = v8f_get(cn, n & 7) * 0.0625f;
    out[tile * 16 + n]         = sp;
    out[BATCH + tile * 16 + n] = sn;
  }
}

// ---------------------------------------------------------------------------
extern "C" void kernel_launch(void* const* d_in, const int* in_sizes, int n_in,
                              void* d_out, int out_size, void* d_ws, size_t ws_size,
                              hipStream_t stream) {
  (void)in_sizes; (void)n_in; (void)out_size; (void)ws_size;
  const int*   users = (const int*)d_in[0];
  const int*   pos   = (const int*)d_in[1];
  const int*   neg   = (const int*)d_in[2];
  const int*   rows  = (const int*)d_in[3];
  const int*   cols  = (const int*)d_in[4];
  const float* vals  = (const float*)d_in[5];
  const float* ue    = (const float*)d_in[6];
  const float* ie    = (const float*)d_in[7];
  float*       out   = (float*)d_out;

  float* acc  = (float*)d_ws;       // 38.4 MB
  float* bufA = acc  + NF;          // 38.4 MB
  float* bufB = bufA + NF;          // 38.4 MB

  const int tpb = 256;
  const int vblocks = (int)((NF / 4 + tpb - 1) / tpb);
  const int sblocks = (int)(((long long)N_EDGES * 16 + tpb - 1) / tpb);

  lgcn_init<<<vblocks, tpb, 0, stream>>>(ue, ie, acc, bufA);

  float* x = bufA;
  float* y = bufB;
  for (int layer = 0; layer < 3; ++layer) {
    hipMemsetAsync(y, 0, (size_t)NF * sizeof(float), stream);
    lgcn_spmm<<<sblocks, tpb, 0, stream>>>(rows, cols, vals, x, y);
    lgcn_add<<<vblocks, tpb, 0, stream>>>(acc, y);
    float* t = x; x = y; y = t;
  }

  const int score_threads = (BATCH / 16) * 32;  // one wave per 16-row tile
  lgcn_score<<<score_threads / tpb, tpb, 0, stream>>>(acc, users, pos, neg, out);
}
